// W8DXLinear_41850161332882
// MI455X (gfx1250) — 8-bit matrix operand order repaired and checked by reading, compiled, not run on hardware
//
#include <hip/hip_runtime.h>

typedef int v8i __attribute__((ext_vector_type(8)));

#define D_IN   4096
#define D_OUT  4096
#define MROWS  8192
#define KOUT   16
#define QF     127.0f

__global__ void k_wquant(const float* __restrict__ W, signed char* __restrict__ wq,
                         float* __restrict__ wmax) {
  __shared__ float red[256];
  const int o = blockIdx.x;
  const int t = threadIdx.x;
  const float* row = W + (size_t)o * D_IN;
  float m = 0.f;
  for (int i = t; i < D_IN; i += 256) m = fmaxf(m, fabsf(row[i]));
  red[t] = m; __syncthreads();
  for (int s = 128; s > 0; s >>= 1) { if (t < s) red[t] = fmaxf(red[t], red[t + s]); __syncthreads(); }
  const float mx = red[0];
  if (t == 0) wmax[o] = mx;
  const float scale = QF / mx;
  signed char* qrow = wq + (size_t)o * D_IN;
  for (int i = t; i < D_IN; i += 256) qrow[i] = (signed char)rintf(row[i] * scale);
}

__global__ void k_colmax(const float* __restrict__ x, float* __restrict__ colmax) {
  const int col = blockIdx.x * 256 + threadIdx.x;
  float m = 0.f;
  for (int r = 0; r < MROWS; ++r) m = fmaxf(m, fabsf(x[(size_t)r * D_IN + col]));
  colmax[col] = m;
}

__global__ void k_topk(const float* __restrict__ colmax, int* __restrict__ idx) {
  __shared__ float sv[D_IN];
  __shared__ float rv[256];
  __shared__ int   ri[256];
  const int t = threadIdx.x;
  for (int i = t; i < D_IN; i += 256) sv[i] = colmax[i];
  __syncthreads();
  for (int k = 0; k < KOUT; ++k) {
    float bv = -1.f; int bi = 0;
    for (int i = t; i < D_IN; i += 256) { float v = sv[i]; if (v > bv) { bv = v; bi = i; } }
    rv[t] = bv; ri[t] = bi; __syncthreads();
    for (int s = 128; s > 0; s >>= 1) {
      if (t < s) {
        float a = rv[t], b = rv[t + s];
        if (b > a || (b == a && ri[t + s] < ri[t])) { rv[t] = b; ri[t] = ri[t + s]; }
      }
      __syncthreads();
    }
    if (t == 0) { idx[k] = ri[0]; sv[ri[0]] = -1.f; }
    __syncthreads();
  }
}

__global__ void k_wunq(const signed char* __restrict__ wq, const float* __restrict__ wmax,
                       const int* __restrict__ idx, float* __restrict__ wunq) {
  const int o = blockIdx.x * 256 + threadIdx.x;
  const float s = wmax[o] * (1.0f / QF);
  for (int k = 0; k < KOUT; ++k)
    wunq[o * KOUT + k] = (float)wq[(size_t)o * D_IN + idx[k]] * s;
}

__global__ void k_xquant(const float* __restrict__ x, const int* __restrict__ idx,
                         signed char* __restrict__ xq, float* __restrict__ xmax,
                         float* __restrict__ xunq) {
  __shared__ int sidx[KOUT];
  __shared__ float red[256];
  const int s = blockIdx.x;
  const int t = threadIdx.x;
  const float* row = x + (size_t)s * D_IN;
  if (t < KOUT) sidx[t] = idx[t];
  __syncthreads();
  if (t < KOUT) xunq[s * KOUT + t] = row[sidx[t]];
  float m = 0.f;
  for (int i = t; i < D_IN; i += 256) {
    bool outl = false;
#pragma unroll
    for (int k = 0; k < KOUT; ++k) outl |= (i == sidx[k]);
    if (!outl) m = fmaxf(m, fabsf(row[i]));
  }
  red[t] = m; __syncthreads();
  for (int r = 128; r > 0; r >>= 1) { if (t < r) red[t] = fmaxf(red[t], red[t + r]); __syncthreads(); }
  const float mx = red[0];
  if (t == 0) xmax[s] = mx;
  const float scale = QF / mx;
  signed char* qrow = xq + (size_t)s * D_IN;
  for (int i = t; i < D_IN; i += 256) {
    bool outl = false;
#pragma unroll
    for (int k = 0; k < KOUT; ++k) outl |= (i == sidx[k]);
    const float v = outl ? 0.f : row[i];
    qrow[i] = (signed char)rintf(v * scale);
  }
}

__global__ __launch_bounds__(256)
void k_gemm(const signed char* __restrict__ xq, const signed char* __restrict__ wq,
            const float* __restrict__ xmax, const float* __restrict__ wmax,
            const float* __restrict__ xunq, const float* __restrict__ wunq,
            const float* __restrict__ bias, float* __restrict__ out) {
  __shared__ uint4 lA[2][1024];
  __shared__ uint4 lB[2][1024];
  const int t    = threadIdx.x;
  const int lane = t & 31;
  const int wave = t >> 5;
  const int hl   = lane >> 4;
  const int ln   = lane & 15;
  const int wm   = wave & 3;
  const int wn   = wave >> 2;
  const int bm0  = blockIdx.y * 128;
  const int bn0  = blockIdx.x * 128;

  const uint4* xq4 = (const uint4*)xq;
  const uint4* wq4 = (const uint4*)wq;

  v8i acc[2][4];
  const v8i vzero = {0, 0, 0, 0, 0, 0, 0, 0};
#pragma unroll
  for (int mt = 0; mt < 2; ++mt)
#pragma unroll
    for (int nt = 0; nt < 4; ++nt) acc[mt][nt] = vzero;

#pragma unroll
  for (int j = 0; j < 4; ++j) {
    const int chunk = j * 256 + t;
    const int row = chunk >> 3;
    const int c   = chunk & 7;
    lA[0][chunk] = xq4[(size_t)(bm0 + row) * 256 + c];
    lB[0][chunk] = wq4[(size_t)(bn0 + row) * 256 + c];
  }
  __syncthreads();

  const int NITER = D_IN / 128;
  for (int it = 0; it < NITER; ++it) {
    const int buf = it & 1;
    uint4 ra[4], rb[4];
    const bool hasNext = (it + 1 < NITER);
    if (hasNext) {
      const int kb16 = (it + 1) * 8;
#pragma unroll
      for (int j = 0; j < 4; ++j) {
        const int chunk = j * 256 + t;
        const int row = chunk >> 3;
        const int c   = chunk & 7;
        ra[j] = xq4[(size_t)(bm0 + row) * 256 + kb16 + c];
        rb[j] = wq4[(size_t)(bn0 + row) * 256 + kb16 + c];
        __builtin_prefetch(&xq4[(size_t)(bm0 + row) * 256 + kb16 + 8 + c], 0, 0);
        __builtin_prefetch(&wq4[(size_t)(bn0 + row) * 256 + kb16 + 8 + c], 0, 0);
      }
    }

    const uint2* lA2 = (const uint2*)lA[buf];
    const uint4* lB4 = lB[buf];
#pragma unroll
    for (int kc = 0; kc < 2; ++kc) {
      v8i bfrag[4];
#pragma unroll
      for (int nt = 0; nt < 4; ++nt) {
        const int n = wn * 64 + nt * 16 + ln;
        const uint4 lo = lB4[n * 8 + kc * 4 + hl];
        const uint4 hi = lB4[n * 8 + kc * 4 + 2 + hl];
        v8i b;
        b[0] = (int)lo.x; b[1] = (int)lo.y; b[2] = (int)lo.z; b[3] = (int)lo.w;
        b[4] = (int)hi.x; b[5] = (int)hi.y; b[6] = (int)hi.z; b[7] = (int)hi.w;
        bfrag[nt] = b;
      }
#pragma unroll
      for (int mt = 0; mt < 2; ++mt) {
        const int rA = wm * 32 + mt * 16 + ln;
        v8i a;
#pragma unroll
        for (int c = 0; c < 4; ++c) {
          const uint2 u = lA2[rA * 16 + kc * 8 + (c >> 1) * 4 + hl * 2 + (c & 1)];
          a[2 * c]     = (int)u.x;
          a[2 * c + 1] = (int)u.y;
        }
#pragma unroll
        for (int nt = 0; nt < 4; ++nt) {
          acc[mt][nt] = __builtin_amdgcn_wmma_i32_16x16x64_iu8(
              true, a, true, bfrag[nt], acc[mt][nt], false, false);
        }
      }
    }

    if (hasNext) {
      const int nbuf = buf ^ 1;
#pragma unroll
      for (int j = 0; j < 4; ++j) {
        const int chunk = j * 256 + t;
        lA[nbuf][chunk] = ra[j];
        lB[nbuf][chunk] = rb[j];
      }
    }
    __syncthreads();
  }

#pragma unroll
  for (int mt = 0; mt < 2; ++mt) {
#pragma unroll
    for (int j = 0; j < 8; ++j) {
      const int srow = bm0 + wm * 32 + mt * 16 + hl * 8 + j;
      const float xm = xmax[srow];
      const float4* xu = (const float4*)(xunq + (size_t)srow * KOUT);
      const float4 xu0 = xu[0], xu1 = xu[1], xu2 = xu[2], xu3 = xu[3];
#pragma unroll
      for (int nt = 0; nt < 4; ++nt) {
        const int o = bn0 + wn * 64 + nt * 16 + ln;
        const float deq = (float)acc[mt][nt][j] / (QF * QF);
        const _Float16 h = (_Float16)deq;
        float r = (float)h * xm * wmax[o];
        const float4* wu = (const float4*)(wunq + (size_t)o * KOUT);
        const float4 wu0 = wu[0], wu1 = wu[1], wu2 = wu[2], wu3 = wu[3];
        float a2 = bias[o];
        a2 += xu0.x * wu0.x + xu0.y * wu0.y + xu0.z * wu0.z + xu0.w * wu0.w;
        a2 += xu1.x * wu1.x + xu1.y * wu1.y + xu1.z * wu1.z + xu1.w * wu1.w;
        a2 += xu2.x * wu2.x + xu2.y * wu2.y + xu2.z * wu2.z + xu2.w * wu2.w;
        a2 += xu3.x * wu3.x + xu3.y * wu3.y + xu3.z * wu3.z + xu3.w * wu3.w;
        out[(size_t)srow * D_OUT + o] = r + a2;
      }
    }
  }
}

extern "C" void kernel_launch(void* const* d_in, const int* in_sizes, int n_in,
                              void* d_out, int out_size, void* d_ws, size_t ws_size,
                              hipStream_t stream) {
  const float* x    = (const float*)d_in[0];
  const float* W    = (const float*)d_in[1];
  const float* bias = (const float*)d_in[2];
  float* out = (float*)d_out;

  char* wsb = (char*)d_ws;
  size_t off = 0;
  auto take = [&](size_t bytes) -> char* {
    char* p = wsb + off;
    off += (bytes + 255) & ~(size_t)255;
    return p;
  };
  signed char* wq   = (signed char*)take((size_t)D_OUT * D_IN);
  signed char* xq   = (signed char*)take((size_t)MROWS * D_IN);
  float* wmax   = (float*)take((size_t)D_OUT * 4);
  float* xmax   = (float*)take((size_t)MROWS * 4);
  float* colmax = (float*)take((size_t)D_IN * 4);
  int*   idx    = (int*)take((size_t)KOUT * 4);
  float* xunq   = (float*)take((size_t)MROWS * KOUT * 4);
  float* wunq   = (float*)take((size_t)D_OUT * KOUT * 4);

  k_wquant<<<D_OUT, 256, 0, stream>>>(W, wq, wmax);
  k_colmax<<<D_IN / 256, 256, 0, stream>>>(x, colmax);
  k_topk<<<1, 256, 0, stream>>>(colmax, idx);
  k_wunq<<<D_OUT / 256, 256, 0, stream>>>(wq, wmax, idx, wunq);
  k_xquant<<<MROWS, 256, 0, stream>>>(x, idx, xq, xmax, xunq);
  dim3 grid(D_OUT / 128, MROWS / 128);
  k_gemm<<<grid, 256, 0, stream>>>(xq, wq, xmax, wmax, xunq, wunq, bias, out);
}
